// PrototypicalReasoningNetwork_82119774699676
// MI455X (gfx1250) — matrix operand order repaired and checked by reading, not compiled, not run on hardware
//
#include <hip/hip_runtime.h>
#include <hip/hip_bf16.h>

// ---------------------------------------------------------------------------
// Prototypical network meta-loss on MI455X (gfx1250, wave32, WMMA).
// Embedding MLP runs as a fused two-stage f16 WMMA GEMM with the hidden
// activation kept entirely in LDS (200KB/block; CDNA5 WGP has 320KB).
// ---------------------------------------------------------------------------

typedef _Float16 v16h __attribute__((ext_vector_type(16)));
typedef _Float16 v8h  __attribute__((ext_vector_type(8)));
typedef float    v8f  __attribute__((ext_vector_type(8)));
typedef float    v4f  __attribute__((ext_vector_type(4)));

#define T_TASKS 16
#define S_SUP   2048
#define Q_QRY   2048
#define DIN     256
#define HID     512
#define DEMB    128
#define NWAY    64

#define LDA_PAD 264   // 256 + 8 halfs -> 528B row stride (16B aligned, bank-shifted)
#define LDH_PAD 520   // 512 + 8 halfs -> 1040B row stride

// ---- WMMA fragment loaders (16-bit A/B, contiguous k = 16h + s order) -------------
// A: lane holds row (lane&15); 16 contiguous K values kb..kb+15,
//    kb = (lane>>4)*16. Caller passes p0 = base + (lane&15)*ld + kb.
__device__ __forceinline__ v16h frag_a_f16(const _Float16* p0) {
  v8h lo = *(const v8h*)(p0);
  v8h hi = *(const v8h*)(p0 + 8);
  return __builtin_shufflevector(lo, hi, 0,1,2,3,4,5,6,7,8,9,10,11,12,13,14,15);
}
// B (stored as [N][K], i.e. W^T row-major): lane holds col (lane&15);
//    16 contiguous K values starting at (lane>>4)*16.
__device__ __forceinline__ v16h frag_b_f16(const _Float16* p0) {
  v8h lo = *(const v8h*)(p0);
  v8h hi = *(const v8h*)(p0 + 8);
  return __builtin_shufflevector(lo, hi, 0,1,2,3,4,5,6,7,8,9,10,11,12,13,14,15);
}

// ---- weight transpose + f32->f16 convert: out[n*K+k] = in[k*N+n] -----------
__global__ void transpose_cvt_kernel(const float* __restrict__ in,
                                     _Float16* __restrict__ out, int K, int N) {
  int i = blockIdx.x * blockDim.x + threadIdx.x;
  if (i >= K * N) return;
  int n = i / K;
  int k = i - n * K;
  out[i] = (_Float16)in[(size_t)k * N + n];
}

// ---- fused 2-layer MLP embedding: E = relu(X@W1 + b1)@W2 + b2 --------------
// One block = 128 rows. Phase0: X tile -> LDS f16. Phase1: H(128x512) -> LDS.
// Phase2: E(128x128) -> global f32. 8 waves, wave tile 32x64, 8 WMMA/k-step.
__global__ __launch_bounds__(256) void embed_fused_kernel(
    const float* __restrict__ X, const _Float16* __restrict__ W1t,
    const float* __restrict__ b1, const _Float16* __restrict__ W2t,
    const float* __restrict__ b2, float* __restrict__ E) {
  extern __shared__ char smem[];
  _Float16* sA = (_Float16*)smem;          // [128][LDA_PAD]
  _Float16* sH = sA + 128 * LDA_PAD;       // [128][LDH_PAD]

  const int tid  = threadIdx.x;
  const int lane = tid & 31;
  const int wid  = tid >> 5;
  const int wm   = wid & 3;                // row group: rows wm*32..wm*32+31
  const int wn   = wid >> 2;               // col group: cols wn*64..wn*64+63
  const size_t rowBase = (size_t)blockIdx.x * 128;

  // Phase 0: stage 128x256 f32 tile as f16 in LDS (float4 loads).
  for (int t = tid; t < (128 * DIN) / 4; t += 256) {
    int e = t * 4;
    int r = e >> 8;          // /256
    int c = e & 255;
    v4f v = *(const v4f*)(X + (rowBase + r) * DIN + c);
    _Float16* d = sA + r * LDA_PAD + c;
    d[0] = (_Float16)v.x; d[1] = (_Float16)v.y;
    d[2] = (_Float16)v.z; d[3] = (_Float16)v.w;
  }
  __syncthreads();

  const int ar = lane & 15;
  const int ak = (lane >> 4) << 4;         // A fragment K sub-offset
  const int bk = (lane >> 4) << 4;         // B fragment K sub-offset
  const int cr = (lane >> 4) << 3;         // C/D row sub-offset

  // Phase 1: H = relu(X @ W1^T + b1), four 128-column chunks of HID=512.
  for (int nc = 0; nc < 4; ++nc) {
    v8f acc[2][4] = {};
    for (int k0 = 0; k0 < DIN; k0 += 32) {
      v16h a[2], b[4];
      #pragma unroll
      for (int i = 0; i < 2; ++i)
        a[i] = frag_a_f16(sA + (wm * 32 + i * 16 + ar) * LDA_PAD + k0 + ak);
      #pragma unroll
      for (int j = 0; j < 4; ++j) {
        int n = nc * 128 + wn * 64 + j * 16 + ar;
        b[j] = frag_b_f16(W1t + (size_t)n * DIN + k0 + bk);
      }
      #pragma unroll
      for (int i = 0; i < 2; ++i)
        #pragma unroll
        for (int j = 0; j < 4; ++j)
          acc[i][j] = __builtin_amdgcn_wmma_f32_16x16x32_f16(
              false, a[i], false, b[j], (short)0, acc[i][j], false, false);
    }
    // Epilogue: bias + relu, write f16 chunk into sH.
    #pragma unroll
    for (int j = 0; j < 4; ++j) {
      int n = nc * 128 + wn * 64 + j * 16 + ar;
      float bias = b1[n];
      #pragma unroll
      for (int i = 0; i < 2; ++i) {
        int r0 = wm * 32 + i * 16 + cr;
        #pragma unroll
        for (int r = 0; r < 8; ++r) {
          float v = acc[i][j][r] + bias;
          sH[(r0 + r) * LDH_PAD + n] = (_Float16)(v > 0.f ? v : 0.f);
        }
      }
    }
  }
  __syncthreads();

  // Phase 2: E = H @ W2^T + b2  (N = DEMB = 128, K = HID = 512).
  v8f acc2[2][4] = {};
  for (int k0 = 0; k0 < HID; k0 += 32) {
    v16h a[2], b[4];
    #pragma unroll
    for (int i = 0; i < 2; ++i)
      a[i] = frag_a_f16(sH + (wm * 32 + i * 16 + ar) * LDH_PAD + k0 + ak);
    #pragma unroll
    for (int j = 0; j < 4; ++j) {
      int n = wn * 64 + j * 16 + ar;
      b[j] = frag_b_f16(W2t + (size_t)n * HID + k0 + bk);
    }
    #pragma unroll
    for (int i = 0; i < 2; ++i)
      #pragma unroll
      for (int j = 0; j < 4; ++j)
        acc2[i][j] = __builtin_amdgcn_wmma_f32_16x16x32_f16(
            false, a[i], false, b[j], (short)0, acc2[i][j], false, false);
  }
  #pragma unroll
  for (int j = 0; j < 4; ++j) {
    int n = wn * 64 + j * 16 + ar;
    float bias = b2[n];
    #pragma unroll
    for (int i = 0; i < 2; ++i) {
      size_t r0 = rowBase + wm * 32 + i * 16 + cr;
      #pragma unroll
      for (int r = 0; r < 8; ++r)
        E[(r0 + r) * DEMB + n] = acc2[i][j][r] + bias;
    }
  }
}

// ---- prototypes: segment-mean over support embeddings ----------------------
// block = (t, c); thread = d. SE per task is 1MB -> L2 resident.
__global__ __launch_bounds__(128) void protos_kernel(
    const float* __restrict__ SE, const int* __restrict__ ids,
    float* __restrict__ protos) {
  const int t = blockIdx.x >> 6;
  const int c = blockIdx.x & 63;
  const int d = threadIdx.x;
  const float* se = SE + (size_t)t * S_SUP * DEMB;
  const int* id = ids + t * S_SUP;
  float sum = 0.f;
  int cnt = 0;
  for (int s = 0; s < S_SUP; ++s) {
    if (id[s] == c) { sum += se[(size_t)s * DEMB + d]; ++cnt; }
  }
  protos[((size_t)(t * NWAY + c)) * DEMB + d] = sum / fmaxf((float)cnt, 1.f);
}

__global__ void init_loss_kernel(double* lossAcc) { lossAcc[0] = 0.0; }

// ---- query pass: cdist -> softmax -> pred -> MSE accumulation --------------
// block = (t, 8 queries), 128 threads; protos cached in LDS (32KB).
__global__ __launch_bounds__(128) void query_loss_kernel(
    const float* __restrict__ QE, const float* __restrict__ protos,
    const float* __restrict__ Y, double* __restrict__ lossAcc) {
  __shared__ float sP[NWAY * DEMB];
  __shared__ float sP2[NWAY];
  __shared__ float sQ[DEMB];
  __shared__ float sL[NWAY];
  __shared__ float sRed[DEMB];
  const int tid = threadIdx.x;
  const int t  = blockIdx.x >> 8;           // / (2048/8)
  const int qb = blockIdx.x & 255;

  const float* P = protos + (size_t)t * NWAY * DEMB;
  for (int i = tid; i < NWAY * DEMB; i += 128) sP[i] = P[i];
  __syncthreads();
  if (tid < NWAY) {
    float s = 0.f;
    for (int d = 0; d < DEMB; ++d) { float v = sP[tid * DEMB + d]; s += v * v; }
    sP2[tid] = s;
  }
  __syncthreads();

  float sqAcc = 0.f;
  for (int qq = 0; qq < 8; ++qq) {
    size_t q = (size_t)qb * 8 + qq;
    sQ[tid] = QE[((size_t)t * Q_QRY + q) * DEMB + tid];
    __syncthreads();
    float v = sQ[tid];
    sRed[tid] = v * v;
    __syncthreads();
    for (int off = 64; off > 0; off >>= 1) {
      if (tid < off) sRed[tid] += sRed[tid + off];
      __syncthreads();
    }
    float q2 = sRed[0];
    if (tid < NWAY) {
      float dot = 0.f;
      const float* pr = sP + tid * DEMB;
      for (int d = 0; d < DEMB; ++d) dot += pr[d] * sQ[d];
      float dist = sqrtf(fmaxf(q2 + sP2[tid] - 2.f * dot, 0.f));
      sL[tid] = -dist;                      // TEMPERATURE = 1.0
    }
    __syncthreads();
    if (tid == 0) {
      float mx = sL[0];
      for (int c = 1; c < NWAY; ++c) mx = fmaxf(mx, sL[c]);
      float ssum = 0.f;
      for (int c = 0; c < NWAY; ++c) { float e = __expf(sL[c] - mx); sL[c] = e; ssum += e; }
      float inv = 1.f / ssum;
      for (int c = 0; c < NWAY; ++c) sL[c] *= inv;
    }
    __syncthreads();
    float pred = 0.f;
    for (int c = 0; c < NWAY; ++c) pred += sL[c] * sP[c * DEMB + tid];
    float diff = pred - Y[((size_t)t * Q_QRY + q) * DEMB + tid];
    sqAcc += diff * diff;
    __syncthreads();
  }
  sRed[tid] = sqAcc;
  __syncthreads();
  for (int off = 64; off > 0; off >>= 1) {
    if (tid < off) sRed[tid] += sRed[tid + off];
    __syncthreads();
  }
  if (tid == 0) atomicAdd(lossAcc, (double)sRed[0]);
}

__global__ void finalize_kernel(const double* __restrict__ lossAcc,
                                float* __restrict__ out) {
  out[0] = (float)(lossAcc[0] / (double)((size_t)T_TASKS * Q_QRY * DEMB));
}

// ---------------------------------------------------------------------------
extern "C" void kernel_launch(void* const* d_in, const int* in_sizes, int n_in,
                              void* d_out, int out_size, void* d_ws, size_t ws_size,
                              hipStream_t stream) {
  const float* support_x  = (const float*)d_in[0];   // [16,2048,256]
  const int*   support_id = (const int*)  d_in[1];   // [16,2048]
  const float* query_x    = (const float*)d_in[2];   // [16,2048,256]
  const float* query_y    = (const float*)d_in[3];   // [16,2048,128]
  const float* W1         = (const float*)d_in[4];   // [256,512]
  const float* b1         = (const float*)d_in[5];   // [512]
  const float* W2         = (const float*)d_in[6];   // [512,128]
  const float* b2         = (const float*)d_in[7];   // [128]
  (void)in_sizes; (void)n_in; (void)out_size; (void)ws_size;

  // Workspace layout (~34.5 MB)
  char* ws = (char*)d_ws;
  const size_t nRowsS = (size_t)T_TASKS * S_SUP;     // 32768
  const size_t nRowsQ = (size_t)T_TASKS * Q_QRY;     // 32768
  float*    E      = (float*)ws;                                      // [65536,128] f32
  size_t    offW1t = (nRowsS + nRowsQ) * DEMB * sizeof(float);        // 33,554,432
  _Float16* W1t    = (_Float16*)(ws + offW1t);                        // [512,256] f16
  size_t    offW2t = offW1t + (size_t)HID * DIN * sizeof(_Float16);
  _Float16* W2t    = (_Float16*)(ws + offW2t);                        // [128,512] f16
  size_t    offP   = offW2t + (size_t)DEMB * HID * sizeof(_Float16);
  float*    protos = (float*)(ws + offP);                             // [16,64,128]
  size_t    offL   = offP + (size_t)T_TASKS * NWAY * DEMB * sizeof(float);
  double*   lossAcc = (double*)(ws + offL);

  float* SE = E;
  float* QE = E + nRowsS * DEMB;

  // 1) weights -> transposed f16
  transpose_cvt_kernel<<<(DIN * HID + 255) / 256, 256, 0, stream>>>(W1, W1t, DIN, HID);
  transpose_cvt_kernel<<<(HID * DEMB + 255) / 256, 256, 0, stream>>>(W2, W2t, HID, DEMB);

  // 2) fused MLP embedding (WMMA), support then query
  const size_t smem = (size_t)(128 * LDA_PAD + 128 * LDH_PAD) * sizeof(_Float16); // 200,704B
  embed_fused_kernel<<<dim3((unsigned)(nRowsS / 128)), 256, smem, stream>>>(
      support_x, W1t, b1, W2t, b2, SE);
  embed_fused_kernel<<<dim3((unsigned)(nRowsQ / 128)), 256, smem, stream>>>(
      query_x, W1t, b1, W2t, b2, QE);

  // 3) prototypes (segment mean)
  protos_kernel<<<T_TASKS * NWAY, 128, 0, stream>>>(SE, support_id, protos);

  // 4) loss accumulation
  init_loss_kernel<<<1, 1, 0, stream>>>(lossAcc);
  query_loss_kernel<<<T_TASKS * (Q_QRY / 8), 128, 0, stream>>>(QE, protos, query_y, lossAcc);
  finalize_kernel<<<1, 1, 0, stream>>>(lossAcc, (float*)d_out);
}
